// Decoder_56994216017940
// MI455X (gfx1250) — matrix operand order repaired and checked by reading, not compiled, not run on hardware
//
#include <hip/hip_runtime.h>
#include <hip/hip_bf16.h>

typedef _Float16 v4h  __attribute__((ext_vector_type(4)));
typedef _Float16 v8h  __attribute__((ext_vector_type(8)));
typedef _Float16 v16h __attribute__((ext_vector_type(16)));
typedef float    v8f  __attribute__((ext_vector_type(8)));

#define WMMA_F32_F16(A, B, C) \
  __builtin_amdgcn_wmma_f32_16x16x32_f16(false, (A), false, (B), (short)0, (C), false, false)

// ---- fragment loaders (per CDNA5 ISA VGPR layouts, wave32) ----
// A 16x32 f16: lane L: M=L%16, kb=(L/16)*8; elems 0..7 -> K=kb..kb+7, 8..15 -> K=kb+16..kb+23
__device__ __forceinline__ v16h fragA_lds(const _Float16* base, int off0, int off1) {
  v8h lo = *(const v8h*)(base + off0);
  v8h hi = *(const v8h*)(base + off1);
  v16h r;
#pragma unroll
  for (int e = 0; e < 8; ++e) { r[e] = lo[e]; r[8 + e] = hi[e]; }
  return r;
}
// B 32x16 f16: lane L: N=L%16, kb=(L/16)*8; elems 0..7 -> K=kb..kb+7, 8..15 -> K=kb+16..kb+23
__device__ __forceinline__ v16h fragB_lds(const _Float16* base, int off) {
  v8h lo = *(const v8h*)(base + off);
  v8h hi = *(const v8h*)(base + off + 16);
  v16h r;
#pragma unroll
  for (int e = 0; e < 8; ++e) { r[e] = lo[e]; r[8 + e] = hi[e]; }
  return r;
}

__device__ __forceinline__ void store4h(_Float16* dst, float4 v) {
  v4h p;
  p[0] = (_Float16)v.x; p[1] = (_Float16)v.y;
  p[2] = (_Float16)v.z; p[3] = (_Float16)v.w;
  *(v4h*)dst = p;
}

// ======================= positional encodings =======================
__global__ void pos1d_kernel(float* __restrict__ dpos, int S, int D) {
  int idx = blockIdx.x * 256 + threadIdx.x;
  if (idx >= S * D) return;
  int p = idx / D, i = idx % D;
  float expo = (float)(2 * (i / 2)) / (float)D;
  float ang = (float)p * powf(10000.0f, -expo);
  dpos[idx] = (i & 1) ? cosf(ang) : sinf(ang);
}

__global__ void pos2d_kernel(float* __restrict__ epos, int D) { // 16x16 grid -> 256 rows
  int idx = blockIdx.x * 256 + threadIdx.x;
  if (idx >= 256 * D) return;
  int p = idx / D, j = idx % D;
  int r = p / 16, c = p % 16;
  int half = D / 2;
  int pos = (j < half) ? r : c;
  int jj  = (j < half) ? j : j - half;
  float expo = (float)(2 * (jj / 2)) / (float)half;
  float ang = (float)pos * powf(10000.0f, -expo);
  epos[idx] = (jj & 1) ? cosf(ang) : sinf(ang);
}

// ======================= embedding gather ===========================
__global__ void embed_kernel(const int* __restrict__ x, const float* __restrict__ emb,
                             float* __restrict__ out, int rows, int D) {
  int idx = blockIdx.x * 256 + threadIdx.x;
  if (idx >= rows * D) return;
  int row = idx / D, d = idx % D;
  out[idx] = emb[(size_t)x[row] * D + d] * 22.62741699796952f; // sqrt(512)
}

// ======================= WMMA GEMM ==================================
// C[M,N] = A[M,K] @ W[K,N] + bias [+ pos[(row&posmask),col]] [relu]
// BM=128 BN=128 BK=32, 256 threads (8 waves), wave -> 32x64 (2x4 WMMA).
// Double-buffered LDS; global loads for tile i+1 issue before WMMAs of tile i.
__global__ __launch_bounds__(256) void gemm_kernel(
    const float* __restrict__ A, const float* __restrict__ W,
    const float* __restrict__ bias, const float* __restrict__ pos,
    float* __restrict__ C, int M, int N, int K, int posmask, int relu) {
  __shared__ _Float16 sA[2][128 * 32];   // [row][k]
  __shared__ _Float16 sBt[2][128 * 32];  // [n][k]
  int t = threadIdx.x;
  int lane = t & 31, wave = t >> 5;
  int wm = wave >> 1, wn = wave & 1;
  int rowbase = blockIdx.y * 128;
  int colbase = blockIdx.x * 128;
  int kb = (lane >> 4) * 8;
  int kb16 = (lane >> 4) * 16;

  v8f acc[2][4] = {};
  float4 ra[4], rw[4];

  auto load_regs = [&](int k0) {
#pragma unroll
    for (int i = 0; i < 4; ++i) { // 128x32 A tile
      int idx = (i * 256 + t) * 4;
      int r = idx >> 5, c = idx & 31;
      ra[i] = *(const float4*)(A + (size_t)(rowbase + r) * K + k0 + c);
    }
#pragma unroll
    for (int i = 0; i < 4; ++i) { // 32x128 W tile
      int idx = (i * 256 + t) * 4;
      int r = idx >> 7, c = idx & 127; // r=k, c=n
      rw[i] = *(const float4*)(W + (size_t)(k0 + r) * N + colbase + c);
    }
  };
  auto store_tile = [&](int buf) {
#pragma unroll
    for (int i = 0; i < 4; ++i) {
      int idx = (i * 256 + t) * 4;
      int r = idx >> 5, c = idx & 31;
      store4h(&sA[buf][r * 32 + c], ra[i]);
    }
#pragma unroll
    for (int i = 0; i < 4; ++i) { // transpose W into [n][k]
      int idx = (i * 256 + t) * 4;
      int r = idx >> 7, c = idx & 127;
      sBt[buf][(c + 0) * 32 + r] = (_Float16)rw[i].x;
      sBt[buf][(c + 1) * 32 + r] = (_Float16)rw[i].y;
      sBt[buf][(c + 2) * 32 + r] = (_Float16)rw[i].z;
      sBt[buf][(c + 3) * 32 + r] = (_Float16)rw[i].w;
    }
  };
  auto compute = [&](int buf) {
    v16h af[2], bf[4];
#pragma unroll
    for (int mi = 0; mi < 2; ++mi) {
      int r = wm * 32 + mi * 16 + (lane & 15);
      af[mi] = fragA_lds(&sA[buf][0], r * 32 + kb, r * 32 + kb + 16);
    }
#pragma unroll
    for (int ni = 0; ni < 4; ++ni) {
      int n = wn * 64 + ni * 16 + (lane & 15);
      bf[ni] = fragB_lds(&sBt[buf][0], n * 32 + kb);
    }
#pragma unroll
    for (int mi = 0; mi < 2; ++mi)
#pragma unroll
      for (int ni = 0; ni < 4; ++ni)
        acc[mi][ni] = WMMA_F32_F16(af[mi], bf[ni], acc[mi][ni]);
  };

  load_regs(0);
  store_tile(0);
  __syncthreads();
  int cur = 0;
  for (int k0 = 32; k0 < K; k0 += 32) {
    load_regs(k0);       // issue next tile's global loads (latency hidden by WMMAs)
    compute(cur);
    store_tile(cur ^ 1); // waits loads, converts, stores to other buffer
    __syncthreads();
    cur ^= 1;
  }
  compute(cur);

  int halfsel = lane >> 4;
#pragma unroll
  for (int mi = 0; mi < 2; ++mi) {
#pragma unroll
    for (int ni = 0; ni < 4; ++ni) {
      int col = colbase + wn * 64 + ni * 16 + (lane & 15);
      float bval = bias ? bias[col] : 0.0f;
#pragma unroll
      for (int r = 0; r < 8; ++r) {
        int row = rowbase + wm * 32 + mi * 16 + r + halfsel * 8;
        float v = acc[mi][ni][r] + bval;
        if (pos) v += pos[(size_t)(row & posmask) * N + col];
        if (relu) v = fmaxf(v, 0.0f);
        C[(size_t)row * N + col] = v;
      }
    }
  }
}

// ======================= flash attention ============================
// Q:[B,SQ,D] K,V:[B,SK,D], head h uses cols h*64..h*64+63. O:[B,SQ,D].
// grid: (SQ/128, H, B); 8 waves x 16 q-rows; 32-key tiles, online softmax.
// Double-buffered K/V staging pipelined against WMMA + softmax.
__global__ __launch_bounds__(256) void flash_kernel(
    const float* __restrict__ Q, const float* __restrict__ Kv,
    const float* __restrict__ Vv, float* __restrict__ O,
    int SQ, int SK, int D, int causal) {
  __shared__ _Float16 sK[2][32 * 64];  // [key][d]
  __shared__ _Float16 sVt[2][64 * 32]; // [d][key]
  __shared__ _Float16 sP[8][16 * 32];  // per-wave P [row][key]

  int t = threadIdx.x, lane = t & 31, wave = t >> 5;
  int h = blockIdx.y, b = blockIdx.z;
  int q0blk = blockIdx.x * 128;
  int q0 = q0blk + wave * 16;
  const float scale = 0.125f; // 1/sqrt(64)
  int Mr = lane & 15, kb = (lane >> 4) * 8, kb16 = (lane >> 4) * 16;
  int halfsel = lane >> 4;

  // Q A-fragments (depth 64 -> two k=32 chunks), vectorized from global
  const float* qp = Q + ((size_t)(b * SQ + q0 + Mr)) * D + h * 64;
  v16h aq[2];
#pragma unroll
  for (int c = 0; c < 2; ++c) {
    float4 x0 = *(const float4*)(qp + c * 32 + kb);
    float4 x1 = *(const float4*)(qp + c * 32 + kb + 4);
    float4 y0 = *(const float4*)(qp + c * 32 + kb + 16);
    float4 y1 = *(const float4*)(qp + c * 32 + kb + 20);
    aq[c][0] = (_Float16)x0.x;  aq[c][1] = (_Float16)x0.y;
    aq[c][2] = (_Float16)x0.z;  aq[c][3] = (_Float16)x0.w;
    aq[c][4] = (_Float16)x1.x;  aq[c][5] = (_Float16)x1.y;
    aq[c][6] = (_Float16)x1.z;  aq[c][7] = (_Float16)x1.w;
    aq[c][8] = (_Float16)y0.x;  aq[c][9] = (_Float16)y0.y;
    aq[c][10] = (_Float16)y0.z; aq[c][11] = (_Float16)y0.w;
    aq[c][12] = (_Float16)y1.x; aq[c][13] = (_Float16)y1.y;
    aq[c][14] = (_Float16)y1.z; aq[c][15] = (_Float16)y1.w;
  }

  v8f o_acc[4] = {};
  float m_r[8], l_r[8];
#pragma unroll
  for (int r = 0; r < 8; ++r) { m_r[r] = -3.0e38f; l_r[r] = 0.0f; }

  float4 rk[2], rv[2];
  auto load_regs = [&](int ks) {
#pragma unroll
    for (int i = 0; i < 2; ++i) {
      int idx = (i * 256 + t) * 4;
      int kk = idx >> 6, d = idx & 63;
      size_t g = ((size_t)(b * SK + ks + kk)) * D + h * 64 + d;
      rk[i] = *(const float4*)(Kv + g);
      rv[i] = *(const float4*)(Vv + g);
    }
  };
  auto store_tile = [&](int buf) {
#pragma unroll
    for (int i = 0; i < 2; ++i) {
      int idx = (i * 256 + t) * 4;
      int kk = idx >> 6, d = idx & 63;
      store4h(&sK[buf][kk * 64 + d], rk[i]);
      sVt[buf][(d + 0) * 32 + kk] = (_Float16)rv[i].x;
      sVt[buf][(d + 1) * 32 + kk] = (_Float16)rv[i].y;
      sVt[buf][(d + 2) * 32 + kk] = (_Float16)rv[i].z;
      sVt[buf][(d + 3) * 32 + kk] = (_Float16)rv[i].w;
    }
  };

  int kmax = causal ? ((q0blk + 128 < SK) ? q0blk + 128 : SK) : SK;
  load_regs(0);
  store_tile(0);
  __syncthreads();
  int cur = 0;
  for (int ks = 0; ks < kmax; ks += 32) {
    bool have_next = (ks + 32 < kmax);
    if (have_next) load_regs(ks + 32); // issue next K/V tile loads early

    if (!(causal && ks > q0 + 15)) {   // wave-uniform skip of fully-masked tiles
      // S = Q @ K^T  (two 16-key sub-tiles, depth split 32+32)
      v8f sfrag[2] = {};
#pragma unroll
      for (int sub = 0; sub < 2; ++sub) {
        int key = sub * 16 + (lane & 15);
#pragma unroll
        for (int c = 0; c < 2; ++c) {
          v16h bk = fragB_lds(&sK[cur][0], key * 64 + c * 32 + kb);
          sfrag[sub] = WMMA_F32_F16(aq[c], bk, sfrag[sub]);
        }
      }
      // scale + causal mask
#pragma unroll
      for (int sub = 0; sub < 2; ++sub) {
        int key = ks + sub * 16 + (lane & 15);
#pragma unroll
        for (int r = 0; r < 8; ++r) {
          int qrow = q0 + r + halfsel * 8;
          float v = sfrag[sub][r] * scale;
          if (causal && key > qrow) v = -1.0e9f;
          sfrag[sub][r] = v;
        }
      }
      // online softmax: row max across 16-lane half
      float fac[8];
#pragma unroll
      for (int r = 0; r < 8; ++r) {
        float mv = fmaxf(sfrag[0][r], sfrag[1][r]);
#pragma unroll
        for (int msk = 8; msk >= 1; msk >>= 1)
          mv = fmaxf(mv, __shfl_xor(mv, msk, 32));
        float nm = fmaxf(m_r[r], mv);
        fac[r] = __expf(m_r[r] - nm);
        m_r[r] = nm;
      }
      // p = exp(s - m); stage to per-wave LDS
#pragma unroll
      for (int sub = 0; sub < 2; ++sub)
#pragma unroll
        for (int r = 0; r < 8; ++r) {
          float p = __expf(sfrag[sub][r] - m_r[r]);
          sfrag[sub][r] = p;
          sP[wave][(r + halfsel * 8) * 32 + sub * 16 + (lane & 15)] = (_Float16)p;
        }
      // row sums, update l
#pragma unroll
      for (int r = 0; r < 8; ++r) {
        float sum = sfrag[0][r] + sfrag[1][r];
#pragma unroll
        for (int msk = 8; msk >= 1; msk >>= 1)
          sum += __shfl_xor(sum, msk, 32);
        l_r[r] = l_r[r] * fac[r] + sum;
      }
      // rescale O, then O += P @ V
#pragma unroll
      for (int nc = 0; nc < 4; ++nc)
#pragma unroll
        for (int r = 0; r < 8; ++r)
          o_acc[nc][r] *= fac[r];
      v16h ap = fragA_lds(&sP[wave][0], Mr * 32 + kb, Mr * 32 + kb + 16);
#pragma unroll
      for (int nc = 0; nc < 4; ++nc) {
        int d = nc * 16 + (lane & 15);
        v16h bv = fragB_lds(&sVt[cur][0], d * 32 + kb);
        o_acc[nc] = WMMA_F32_F16(ap, bv, o_acc[nc]);
      }
    }

    if (have_next) store_tile(cur ^ 1);
    __syncthreads();
    cur ^= 1;
  }

  // normalize and write
#pragma unroll
  for (int nc = 0; nc < 4; ++nc) {
    int d = h * 64 + nc * 16 + (lane & 15);
#pragma unroll
    for (int r = 0; r < 8; ++r) {
      int qrow = q0 + r + halfsel * 8;
      O[((size_t)(b * SQ + qrow)) * D + d] = o_acc[nc][r] / l_r[r];
    }
  }
}

// =============== residual + layernorm ===============
__global__ __launch_bounds__(256) void ln_kernel(
    const float* __restrict__ x, const float* __restrict__ res,
    const float* __restrict__ g, const float* __restrict__ bta,
    float* __restrict__ out, int D) {
  int row = blockIdx.x, t = threadIdx.x;
  const float* xr = x + (size_t)row * D;
  const float* rr = res + (size_t)row * D;
  float* orow = out + (size_t)row * D;
  __shared__ float red[256];
  float local = 0.0f;
  for (int i = t; i < D; i += 256) local += xr[i] + rr[i];
  red[t] = local; __syncthreads();
  for (int s = 128; s > 0; s >>= 1) { if (t < s) red[t] += red[t + s]; __syncthreads(); }
  float mean = red[0] / (float)D;
  __syncthreads();
  float lv = 0.0f;
  for (int i = t; i < D; i += 256) { float d = xr[i] + rr[i] - mean; lv += d * d; }
  red[t] = lv; __syncthreads();
  for (int s = 128; s > 0; s >>= 1) { if (t < s) red[t] += red[t + s]; __syncthreads(); }
  float inv = rsqrtf(red[0] / (float)D + 1e-6f);
  for (int i = t; i < D; i += 256)
    orow[i] = (xr[i] + rr[i] - mean) * inv * g[i] + bta[i];
}

// =============== row softmax (in place) ===============
__global__ __launch_bounds__(256) void softmax_kernel(float* __restrict__ x, int N) {
  int row = blockIdx.x, t = threadIdx.x;
  float* xr = x + (size_t)row * N;
  __shared__ float red[256];
  float mv = -3.0e38f;
  for (int i = t; i < N; i += 256) mv = fmaxf(mv, xr[i]);
  red[t] = mv; __syncthreads();
  for (int s = 128; s > 0; s >>= 1) { if (t < s) red[t] = fmaxf(red[t], red[t + s]); __syncthreads(); }
  float m = red[0]; __syncthreads();
  float sum = 0.0f;
  for (int i = t; i < N; i += 256) { float e = __expf(xr[i] - m); xr[i] = e; sum += e; }
  red[t] = sum; __syncthreads();
  for (int s = 128; s > 0; s >>= 1) { if (t < s) red[t] += red[t + s]; __syncthreads(); }
  float inv = 1.0f / red[0];
  for (int i = t; i < N; i += 256) xr[i] *= inv;
}

// ======================= driver =======================
extern "C" void kernel_launch(void* const* d_in, const int* in_sizes, int n_in,
                              void* d_out, int out_size, void* d_ws, size_t ws_size,
                              hipStream_t stream) {
  const int B = 16, S = 512, D = 512, F = 2048, L = 4, E = 256, H = 8;
  const int M = B * S;        // 8192
  const int ME = B * E;       // 4096

  const int*   x    = (const int*)d_in[0];
  const float* enc  = (const float*)d_in[1];
  // d_in[2] = look_ahead_mask (causal; computed analytically, unused)
  const float* emb  = (const float*)d_in[3];
  const float* m1w  = (const float*)d_in[4];
  const float* m1b  = (const float*)d_in[5];
  const float* m2w  = (const float*)d_in[6];
  const float* m2b  = (const float*)d_in[7];
  const float* fw1  = (const float*)d_in[8];
  const float* fb1  = (const float*)d_in[9];
  const float* fw2  = (const float*)d_in[10];
  const float* fb2  = (const float*)d_in[11];
  const float* lng  = (const float*)d_in[12];
  const float* lnb  = (const float*)d_in[13];
  const float* outw = (const float*)d_in[14];
  const float* outb = (const float*)d_in[15];
  float* logits = (float*)d_out;

  float* ws = (float*)d_ws;
  const size_t SZ = (size_t)M * D; // 4,194,304 floats
  float* hbuf = ws;
  float* qb   = hbuf + SZ;
  float* kbuf = qb   + SZ;
  float* vbuf = kbuf + SZ;
  float* aob  = vbuf + SZ;
  float* t1   = aob  + SZ;
  float* o1   = t1   + SZ;
  float* o2   = o1   + SZ;
  float* ffn  = o2   + SZ;                 // M*F floats
  float* dpos = ffn  + (size_t)M * F;      // S*D
  float* epos = dpos + (size_t)S * D;      // 256*D

  auto gemm = [&](const float* A, const float* W, const float* bias, const float* pos,
                  float* C, int Mm, int Nn, int Kk, int posmask, int relu) {
    dim3 g(Nn / 128, Mm / 128);
    hipLaunchKernelGGL(gemm_kernel, g, dim3(256), 0, stream,
                       A, W, bias, pos, C, Mm, Nn, Kk, posmask, relu);
  };

  hipLaunchKernelGGL(pos1d_kernel, dim3((S * D + 255) / 256), dim3(256), 0, stream, dpos, S, D);
  hipLaunchKernelGGL(pos2d_kernel, dim3((256 * D + 255) / 256), dim3(256), 0, stream, epos, D);
  hipLaunchKernelGGL(embed_kernel, dim3((M * D + 255) / 256), dim3(256), 0, stream, x, emb, hbuf, M, D);

  for (int l = 0; l < L; ++l) {
    const float* W1 = m1w + (size_t)l * 4 * D * D;
    const float* B1 = m1b + (size_t)l * 4 * D;
    const float* W2 = m2w + (size_t)l * 4 * D * D;
    const float* B2 = m2b + (size_t)l * 4 * D;

    // ---- self-attention ----
    gemm(hbuf, W1 + 0 * (size_t)D * D, B1 + 0 * D, dpos, qb,   M, D, D, S - 1, 0);
    gemm(hbuf, W1 + 1 * (size_t)D * D, B1 + 1 * D, dpos, kbuf, M, D, D, S - 1, 0);
    gemm(hbuf, W1 + 2 * (size_t)D * D, B1 + 2 * D, nullptr, vbuf, M, D, D, 0, 0);
    hipLaunchKernelGGL(flash_kernel, dim3(S / 128, H, B), dim3(256), 0, stream,
                       qb, kbuf, vbuf, aob, S, S, D, 1);
    gemm(aob, W1 + 3 * (size_t)D * D, B1 + 3 * D, nullptr, t1, M, D, D, 0, 0);
    hipLaunchKernelGGL(ln_kernel, dim3(M), dim3(256), 0, stream,
                       t1, hbuf, lng + (size_t)(l * 3 + 0) * D, lnb + (size_t)(l * 3 + 0) * D, o1, D);

    // ---- cross-attention (K/V from encoder, 2D pos on K) ----
    gemm(o1,  W2 + 0 * (size_t)D * D, B2 + 0 * D, dpos, qb,   M,  D, D, S - 1, 0);
    gemm(enc, W2 + 1 * (size_t)D * D, B2 + 1 * D, epos, kbuf, ME, D, D, E - 1, 0);
    gemm(enc, W2 + 2 * (size_t)D * D, B2 + 2 * D, nullptr, vbuf, ME, D, D, 0, 0);
    hipLaunchKernelGGL(flash_kernel, dim3(S / 128, H, B), dim3(256), 0, stream,
                       qb, kbuf, vbuf, aob, S, E, D, 0);
    gemm(aob, W2 + 3 * (size_t)D * D, B2 + 3 * D, nullptr, t1, M, D, D, 0, 0);
    hipLaunchKernelGGL(ln_kernel, dim3(M), dim3(256), 0, stream,
                       t1, o1, lng + (size_t)(l * 3 + 1) * D, lnb + (size_t)(l * 3 + 1) * D, o2, D);

    // ---- FFN ----
    gemm(o2,  fw1 + (size_t)l * D * F, fb1 + (size_t)l * F, nullptr, ffn, M, F, D, 0, 1);
    gemm(ffn, fw2 + (size_t)l * F * D, fb2 + (size_t)l * D, nullptr, t1,  M, D, F, 0, 0);
    hipLaunchKernelGGL(ln_kernel, dim3(M), dim3(256), 0, stream,
                       t1, o2, lng + (size_t)(l * 3 + 2) * D, lnb + (size_t)(l * 3 + 2) * D, hbuf, D);
  }

  // final projection + softmax over vocab
  gemm(hbuf, outw, outb, nullptr, logits, M, 2048, D, 0, 0);
  hipLaunchKernelGGL(softmax_kernel, dim3(M), dim3(256), 0, stream, logits, 2048);
}
